// TernaryLinear_30159260352826
// MI455X (gfx1250) — matrix operand order repaired and checked by reading, not compiled, not run on hardware
//
#include <hip/hip_runtime.h>

// ---------------------------------------------------------------------------
// TernaryLinear on MI455X (gfx1250, wave32, WMMA)
//   pass 1: ternarize w1,w2 -> bf16 w_t in d_ws; convert x -> bf16 in d_ws
//   pass 2: bf16 WMMA GEMM (f32 accum) + bias, double-buffered LDS with
//           async global->LDS staging (ASYNCcnt pipelined), NT f32 output
// ---------------------------------------------------------------------------

typedef __attribute__((ext_vector_type(16))) __bf16 v16bf;
typedef __attribute__((ext_vector_type(8)))  __bf16 v8bf;
typedef __attribute__((ext_vector_type(8)))  float  v8f;
typedef __attribute__((ext_vector_type(4)))  float  vf4;

#define IN_F   4096
#define OUT_F  16384
#define TOKENS 4096

#define BM  128
#define BN  128
#define BK  64        // two WMMA K-steps per staged tile
#define BKP 72        // padded LDS row stride (bf16): 144B -> conflict-friendly
#define WM  64        // per-wave tile M  (4 frags)
#define WN  32        // per-wave tile N  (2 frags)

#if __has_builtin(__builtin_amdgcn_global_load_async_to_lds_b128) && \
    __has_builtin(__builtin_amdgcn_s_wait_asynccnt)
#define USE_ASYNC_LDS 1
#else
#define USE_ASYNC_LDS 0
#endif

#if USE_ASYNC_LDS
// builtin signature (from hipcc diagnostic): pointers to int4 vectors,
// global source in addrspace(1), LDS destination in addrspace(3)
typedef int v4i __attribute__((vector_size(16)));
typedef __attribute__((address_space(1))) v4i gv4i;   // global int4
typedef __attribute__((address_space(3))) v4i sv4i;   // LDS int4
#endif

// ---------------------------- prep kernels ---------------------------------

__device__ __forceinline__ v8bf cvt8(vf4 a, vf4 b) {
  v8bf r;
#pragma unroll
  for (int j = 0; j < 4; ++j) {
    r[j]     = (__bf16)a[j];
    r[j + 4] = (__bf16)b[j];
  }
  return r;
}

__device__ __forceinline__ vf4 sgn4(vf4 v) {
  vf4 r;
#pragma unroll
  for (int j = 0; j < 4; ++j)
    r[j] = (v[j] > 0.f ? 1.f : 0.f) - (v[j] < 0.f ? 1.f : 0.f);
  return r;
}

__global__ void cvt_x_kernel(const float* __restrict__ x,
                             __bf16* __restrict__ xb, int n8) {
  int c = blockIdx.x * blockDim.x + threadIdx.x;
  if (c >= n8) return;
  const vf4* p = (const vf4*)x + 2 * (size_t)c;
  vf4 a = __builtin_nontemporal_load(p);
  vf4 b = __builtin_nontemporal_load(p + 1);
  __builtin_nontemporal_store(cvt8(a, b), (v8bf*)xb + c);
}

__global__ void ternarize_kernel(const float* __restrict__ w1,
                                 const float* __restrict__ w2,
                                 __bf16* __restrict__ wt, int n8) {
  int c = blockIdx.x * blockDim.x + threadIdx.x;
  if (c >= n8) return;
  const vf4* p1 = (const vf4*)w1 + 2 * (size_t)c;
  const vf4* p2 = (const vf4*)w2 + 2 * (size_t)c;
  vf4 a1 = __builtin_nontemporal_load(p1);
  vf4 b1 = __builtin_nontemporal_load(p1 + 1);
  vf4 a2 = __builtin_nontemporal_load(p2);
  vf4 b2 = __builtin_nontemporal_load(p2 + 1);
  vf4 t0 = (sgn4(a1) + sgn4(a2)) * 0.5f;
  vf4 t1 = (sgn4(b1) + sgn4(b2)) * 0.5f;
  __builtin_nontemporal_store(cvt8(t0, t1), (v8bf*)wt + c);
}

// ------------------------------ GEMM ---------------------------------------
// out[t][o] = sum_i X[t][i] * W[o][i] + bias[o]
// 256 threads = 8 waves (2 x 4); block tile BM x BN; staged K-tile BK=64.

__device__ __forceinline__ void compute_tile(const __bf16 (*Xs)[BKP],
                                             const __bf16 (*Wt)[BKP],
                                             v8f acc[4][2],
                                             int wm, int wn, int lm, int lh) {
#pragma unroll
  for (int ks = 0; ks < BK; ks += 32) {
    // A fragments (ISA 7.12.2 16-bit A layout): lane<16 holds K{0..7,16..23},
    // lane>=16 holds K{8..15,24..31} of row M=lane%16
    v16bf af[4];
#pragma unroll
    for (int r = 0; r < 4; ++r) {
      const __bf16* base = &Xs[wm * WM + r * 16 + lm][ks];
      v8bf lo = *(const v8bf*)(base + lh * 8);
      v8bf hi = *(const v8bf*)(base + 16 + lh * 8);
      af[r] = __builtin_shufflevector(lo, hi, 0, 1, 2, 3, 4, 5, 6, 7,
                                      8, 9, 10, 11, 12, 13, 14, 15);
    }
    // B fragments: same K order as A: lane<16 holds K{0..7,16..23}, lane>=16 K{8..15,24..31} for N=lane%16
    v16bf bfr[2];
#pragma unroll
    for (int c = 0; c < 2; ++c) {
      const __bf16* base = &Wt[wn * WN + c * 16 + lm][ks];
      v8bf lo = *(const v8bf*)(base + lh * 8);
      v8bf hi = *(const v8bf*)(base + 16 + lh * 8);
      bfr[c] = __builtin_shufflevector(lo, hi, 0, 1, 2, 3, 4, 5, 6, 7,
                                       8, 9, 10, 11, 12, 13, 14, 15);
    }
#pragma unroll
    for (int r = 0; r < 4; ++r)
#pragma unroll
      for (int c = 0; c < 2; ++c)
        acc[r][c] = __builtin_amdgcn_wmma_f32_16x16x32_bf16(
            false, af[r], false, bfr[c], (short)0, acc[r][c], false, false);
  }
}

__global__ __launch_bounds__(256)
void ternary_gemm_kernel(const __bf16* __restrict__ X,   // [TOKENS][IN_F]
                         const __bf16* __restrict__ W,   // [OUT_F][IN_F]
                         const float*  __restrict__ bias,
                         float* __restrict__ out) {      // [TOKENS][OUT_F]
#if USE_ASYNC_LDS
  __shared__ __bf16 Xs[2][BM][BKP];   // 2 x 18 KB
  __shared__ __bf16 Wt[2][BN][BKP];   // 2 x 18 KB   (72 KB total, WGP has 320)
#else
  __shared__ __bf16 Xs[1][BM][BKP];
  __shared__ __bf16 Wt[1][BN][BKP];
#endif

  const int tid  = threadIdx.x;
  const int lane = tid & 31;
  const int wave = tid >> 5;
  const int wm   = wave >> 2;         // 0..1
  const int wn   = wave & 3;          // 0..3
  const int lm   = lane & 15;
  const int lh   = lane >> 4;

  const int m0 = blockIdx.y * BM;
  const int n0 = blockIdx.x * BN;

  v8f acc[4][2] = {};
  const int NT = IN_F / BK;           // 64 staged tiles

#if USE_ASYNC_LDS
  // ---- async double-buffered pipeline: 8 b128 async loads/thread/tile ----
  // chunk c in [0,1024): row = c/8, kcol = (c%8)*8  (per matrix)
#define STAGE_ASYNC(buf, k0)                                                  \
  do {                                                                        \
    _Pragma("unroll")                                                         \
    for (int t = 0; t < 4; ++t) {                                             \
      int c   = tid + t * 256;                                                \
      int row = c >> 3;                                                       \
      int kc  = (c & 7) * 8;                                                  \
      __builtin_amdgcn_global_load_async_to_lds_b128(                         \
          (gv4i*)(void*)(X + (size_t)(m0 + row) * IN_F + (k0) + kc),          \
          (sv4i*)(void*)&Xs[(buf)][row][kc], 0, 0);                           \
      __builtin_amdgcn_global_load_async_to_lds_b128(                         \
          (gv4i*)(void*)(W + (size_t)(n0 + row) * IN_F + (k0) + kc),          \
          (sv4i*)(void*)&Wt[(buf)][row][kc], 0, 0);                           \
    }                                                                         \
  } while (0)

  STAGE_ASYNC(0, 0);
  for (int i = 0; i < NT; ++i) {
    const int cur = i & 1;
    if (i + 1 < NT) {
      STAGE_ASYNC(cur ^ 1, (i + 1) * BK);
      if (i + 2 < NT) {      // L2 warm-up two tiles ahead
        int row = tid & 127;
        __builtin_prefetch(X + (size_t)(m0 + row) * IN_F + (i + 2) * BK, 0, 1);
        __builtin_prefetch(W + (size_t)(n0 + row) * IN_F + (i + 2) * BK, 0, 1);
      }
      __builtin_amdgcn_s_wait_asynccnt(8);   // tile i complete (in-order)
    } else {
      __builtin_amdgcn_s_wait_asynccnt(0);
    }
    __syncthreads();                         // tile i visible to all waves
    compute_tile(Xs[cur], Wt[cur], acc, wm, wn, lm, lh);
    __syncthreads();                         // done reading before overwrite
  }
#else
  // ---- fallback: synchronous staging, single buffer ----
  for (int i = 0; i < NT; ++i) {
    const int k0 = i * BK;
    __syncthreads();
#pragma unroll
    for (int t = 0; t < 4; ++t) {
      int c   = tid + t * 256;
      int row = c >> 3;
      int kc  = (c & 7) * 8;
      *(v8bf*)&Xs[0][row][kc] =
          *(const v8bf*)(X + (size_t)(m0 + row) * IN_F + k0 + kc);
      *(v8bf*)&Wt[0][row][kc] =
          *(const v8bf*)(W + (size_t)(n0 + row) * IN_F + k0 + kc);
    }
    if (i + 1 < NT) {
      int row = tid & 127;
      __builtin_prefetch(X + (size_t)(m0 + row) * IN_F + k0 + BK, 0, 1);
      __builtin_prefetch(W + (size_t)(n0 + row) * IN_F + k0 + BK, 0, 1);
    }
    __syncthreads();
    compute_tile(Xs[0], Wt[0], acc, wm, wn, lm, lh);
  }
#endif

  // epilogue: C/D layout — VGPR v holds M = v + 8*lh, N = lane%16
#pragma unroll
  for (int r = 0; r < 4; ++r) {
#pragma unroll
    for (int c = 0; c < 2; ++c) {
      int   col = n0 + wn * WN + c * 16 + lm;
      float bv  = bias[col];
      float* orow =
          out + (size_t)(m0 + wm * WM + r * 16 + lh * 8) * OUT_F + col;
#pragma unroll
      for (int v = 0; v < 8; ++v)
        __builtin_nontemporal_store(acc[r][c][v] + bv,
                                    orow + (size_t)v * OUT_F);
    }
  }
}

// ------------------------------ launch -------------------------------------

extern "C" void kernel_launch(void* const* d_in, const int* in_sizes, int n_in,
                              void* d_out, int out_size, void* d_ws,
                              size_t ws_size, hipStream_t stream) {
  const float* x    = (const float*)d_in[0];   // [TOKENS][IN_F]
  const float* w1   = (const float*)d_in[1];   // [OUT_F][IN_F]
  const float* w2   = (const float*)d_in[2];   // [OUT_F][IN_F]
  const float* bias = (const float*)d_in[3];   // [OUT_F]
  float* out = (float*)d_out;                  // [TOKENS][OUT_F]

  // workspace: x_bf16 (32 MB) then w_t bf16 (128 MB)
  __bf16* xb = (__bf16*)d_ws;
  __bf16* wt = (__bf16*)((char*)d_ws + (size_t)TOKENS * IN_F * sizeof(__bf16));

  {
    int n8 = (TOKENS * IN_F) / 8;
    cvt_x_kernel<<<(n8 + 255) / 256, 256, 0, stream>>>(x, xb, n8);
  }
  {
    int n8 = (int)(((size_t)OUT_F * IN_F) / 8);
    ternarize_kernel<<<(n8 + 255) / 256, 256, 0, stream>>>(w1, w2, wt, n8);
  }

  dim3 grid(OUT_F / BN, TOKENS / BM);
  ternary_gemm_kernel<<<grid, 256, 0, stream>>>(xb, wt, bias, out);
}
